// NeuralTextCodec_75445395522050
// MI455X (gfx1250) — compile-verified, matrix operand order checked by reading, not run on hardware
//
#include <hip/hip_runtime.h>

typedef __attribute__((ext_vector_type(16))) _Float16 v16h;
typedef __attribute__((ext_vector_type(8)))  float    v8f;

#define DEVINL static __device__ __forceinline__

namespace {
constexpr int Bn = 256, Tn = 64, En = 128, Hn = 256, Ln = 128, Kc = 16, Vocab = 8000;
constexpr int NG = 4 * Hn;        // 1024 gate width
constexpr int LSTM_WGS = 32;      // persistent workgroups for recurrent kernels
constexpr size_t LOGITS_N  = (size_t)Bn * Tn * Vocab;   // 131,072,000
constexpr size_t MEAN_OFF  = LOGITS_N;
constexpr size_t LOGVAR_OFF = MEAN_OFF + (size_t)Bn * Ln;
constexpr size_t VQ_OFF    = LOGVAR_OFF + (size_t)Bn * Ln;
constexpr size_t IDX_OFF   = VQ_OFF + 1;
}

// ---------------------------------------------------------------------------
// WMMA fragment helpers (CDNA5 v_wmma_f32_16x16x32_f16, ISA 7.12.2 layouts)
// A (16x32, M x K): lane%16 = M, lane>=16 selects K+8 half; vregs 0-3 hold
// K pairs {kb+0..7}, vregs 4-7 hold {kb+16..23}. B (32x16, K x N) built from
// W[N,K] row-major with lane%16 = N and identical K mapping (B[k][n]=W[n][k]).
// Each fragment is two global_load_b128.
// ---------------------------------------------------------------------------
DEVINL v16h load_frag(const _Float16* __restrict__ base, int ld, int r0, int k0, int lane) {
  int r  = r0 + (lane & 15);
  int kb = k0 + ((lane & 16) ? 8 : 0);
  const _Float16* p = base + (size_t)r * ld + kb;
  v16h f;
#pragma unroll
  for (int j = 0; j < 8; ++j) f[j] = p[j];
#pragma unroll
  for (int j = 0; j < 8; ++j) f[8 + j] = p[16 + j];
  return f;
}

DEVINL v8f wmma_f16(v16h a, v16h b, v8f c) {
  return __builtin_amdgcn_wmma_f32_16x16x32_f16(false, a, false, b, (short)0, c, false, false);
}

DEVINL float sigmoidf_(float x) { return 1.0f / (1.0f + __expf(-x)); }

// ---------------------------------------------------------------------------
// Grid-wide generation barrier (persistent LSTM kernels). Deterministic;
// counters live in d_ws and are re-zeroed by init_kernel every launch.
// ---------------------------------------------------------------------------
DEVINL void grid_barrier(unsigned* gen, unsigned* cnt, unsigned nwg) {
  __threadfence();
  __syncthreads();
  if (threadIdx.x == 0) {
    unsigned g = __hip_atomic_load(gen, __ATOMIC_ACQUIRE, __HIP_MEMORY_SCOPE_AGENT);
    unsigned prev = __hip_atomic_fetch_add(cnt, 1u, __ATOMIC_ACQ_REL, __HIP_MEMORY_SCOPE_AGENT);
    if (prev == nwg - 1) {
      __hip_atomic_store(cnt, 0u, __ATOMIC_RELAXED, __HIP_MEMORY_SCOPE_AGENT);
      __hip_atomic_fetch_add(gen, 1u, __ATOMIC_ACQ_REL, __HIP_MEMORY_SCOPE_AGENT);
    } else {
      while (__hip_atomic_load(gen, __ATOMIC_ACQUIRE, __HIP_MEMORY_SCOPE_AGENT) == g)
        __builtin_amdgcn_s_sleep(1);
    }
  }
  __syncthreads();
  __threadfence();
}

__global__ void init_kernel(unsigned* bar) {
  if (threadIdx.x < 2) bar[threadIdx.x] = 0u;
}

__global__ void cast_f16_kernel(const float* __restrict__ src, _Float16* __restrict__ dst, int n) {
  int i = blockIdx.x * 256 + threadIdx.x;
  if (i < n) dst[i] = (_Float16)src[i];
}

// emb[t][b][e] = tok_emb[x[b][t]][e]  (f32 -> f16)
__global__ void embed_kernel(const int* __restrict__ x, const float* __restrict__ tok_emb,
                             _Float16* __restrict__ emb) {
  size_t i = (size_t)blockIdx.x * 256 + threadIdx.x;
  if (i >= (size_t)Tn * Bn * En) return;
  int e = (int)(i % En);
  size_t tb = i / En;
  int b = (int)(tb % Bn), t = (int)(tb / Bn);
  int tok = x[b * Tn + t];
  emb[i] = (_Float16)tok_emb[(size_t)tok * En + e];
}

// ---------------------------------------------------------------------------
// Generic WMMA GEMM: C[M,N] = A[M,K] @ W[N,K]^T + bias.  Requires K % 64 == 0.
// 8 waves/block arranged 4(M) x 2(N); each wave register-blocks a 32x32
// footprint (2x2 WMMA tiles: 4 fragments -> 4 WMMAs). The K-loop is unrolled
// 2x with ping-pong fragment buffers: no register rotation movs, loads of the
// next 32-chunk are always in flight behind the current chunk's WMMAs.
// Block tile = 128(M) x 64(N). remapB!=0 remaps output row m=t*remapB+b ->
// b*(M/remapB)+t (for the [T,B]->[B,T] logits layout).
// ---------------------------------------------------------------------------
__global__ __launch_bounds__(256) void gemm_kernel(
    const _Float16* __restrict__ A, const _Float16* __restrict__ W,
    const float* __restrict__ bias, float* __restrict__ Cf, _Float16* __restrict__ Ch,
    int M, int N, int K, long ldc, int remapB) {
  int lane = threadIdx.x & 31;
  int wave = threadIdx.x >> 5;                  // 0..7
  int m0 = blockIdx.y * 128 + (wave >> 1) * 32; // wave: 32 rows
  int n0 = blockIdx.x * 64 + (wave & 1) * 32;   // wave: 32 cols
  if (m0 >= M || n0 >= N) return;

  v8f acc00 = {}, acc01 = {}, acc10 = {}, acc11 = {};
  // buffer A <- chunk 0
  v16h a0 = load_frag(A, K, m0,      0, lane);
  v16h a1 = load_frag(A, K, m0 + 16, 0, lane);
  v16h b0 = load_frag(W, K, n0,      0, lane);
  v16h b1 = load_frag(W, K, n0 + 16, 0, lane);

  for (int k0 = 0; k0 < K; k0 += 64) {
    // buffer B <- chunk k0+32 (always exists: K % 64 == 0)
    v16h a0n = load_frag(A, K, m0,      k0 + 32, lane);
    v16h a1n = load_frag(A, K, m0 + 16, k0 + 32, lane);
    v16h b0n = load_frag(W, K, n0,      k0 + 32, lane);
    v16h b1n = load_frag(W, K, n0 + 16, k0 + 32, lane);

    acc00 = wmma_f16(a0, b0, acc00);
    acc01 = wmma_f16(a0, b1, acc01);
    acc10 = wmma_f16(a1, b0, acc10);
    acc11 = wmma_f16(a1, b1, acc11);

    if (k0 + 64 < K) {
      __builtin_prefetch(A + (size_t)m0 * K + k0 + 96, 0, 3);
      __builtin_prefetch(W + (size_t)n0 * K + k0 + 96, 0, 3);
      // buffer A <- chunk k0+64 (in-place; never read when stale)
      a0 = load_frag(A, K, m0,      k0 + 64, lane);
      a1 = load_frag(A, K, m0 + 16, k0 + 64, lane);
      b0 = load_frag(W, K, n0,      k0 + 64, lane);
      b1 = load_frag(W, K, n0 + 16, k0 + 64, lane);
    }
    acc00 = wmma_f16(a0n, b0n, acc00);
    acc01 = wmma_f16(a0n, b1n, acc01);
    acc10 = wmma_f16(a1n, b0n, acc10);
    acc11 = wmma_f16(a1n, b1n, acc11);
  }

  v8f accs[2][2] = {{acc00, acc01}, {acc10, acc11}};
#pragma unroll
  for (int mi = 0; mi < 2; ++mi) {
#pragma unroll
    for (int ni = 0; ni < 2; ++ni) {
      int n = n0 + ni * 16 + (lane & 15);
      float bv = bias ? bias[n] : 0.0f;
      int mbase = m0 + mi * 16 + ((lane & 16) ? 8 : 0);
#pragma unroll
      for (int j = 0; j < 8; ++j) {
        int m = mbase + j;
        size_t row = (size_t)m;
        if (remapB) {
          int b = m % remapB, t = m / remapB;
          row = (size_t)b * (M / remapB) + t;
        }
        float v = accs[mi][ni][j] + bv;
        if (Cf) Cf[row * ldc + n] = v;
        if (Ch) Ch[row * ldc + n] = (_Float16)v;
      }
    }
  }
}

// ---------------------------------------------------------------------------
// Persistent LSTM: 32 WGs x 8 waves. Per step: gates[B,4H] = xp[t] + h@Whh^T
// (256 waves x one M-tile x four N-tiles, ping-pong pipelined K-loop),
// grid barrier, fused i/f/g/o elementwise update, grid barrier.
// xp_tstride=0 for the decoder (broadcast input); reverse=1 reads xp
// time-reversed (bwd encoder).
// ---------------------------------------------------------------------------
__global__ __launch_bounds__(256) void lstm_kernel(
    const _Float16* __restrict__ xp, long xp_tstride, int reverse,
    const _Float16* __restrict__ Whh,
    _Float16* __restrict__ h, float* __restrict__ c, float* __restrict__ gates,
    _Float16* __restrict__ hs,          // [T,B,H] f16 or null
    _Float16* __restrict__ hT, int ldT, // final h (f16) or null
    unsigned* __restrict__ bar) {
  unsigned* gen = bar;
  unsigned* cnt = bar + 1;
  const int tid  = blockIdx.x * 256 + threadIdx.x;         // 0..8191
  const int lane = threadIdx.x & 31;
  const int w    = (blockIdx.x << 3) | (threadIdx.x >> 5); // global wave 0..255
  const int mt   = w & 15;   // batch tile
  const int ngi  = w >> 4;   // gate-tile group: n-tiles ngi*4 .. ngi*4+3

  for (int e = tid; e < Bn * Hn; e += LSTM_WGS * 256) { h[e] = (_Float16)0.0f; c[e] = 0.0f; }
  grid_barrier(gen, cnt, LSTM_WGS);

  for (int t = 0; t < Tn; ++t) {
    const int trow = reverse ? (Tn - 1 - t) : t;
    const _Float16* xpt = xp + (size_t)trow * xp_tstride;

    v8f acc[4] = {{}, {}, {}, {}};
    v16h a = load_frag(h, Hn, mt * 16, 0, lane);
    v16h b[4];
#pragma unroll
    for (int q = 0; q < 4; ++q) b[q] = load_frag(Whh, Hn, (ngi * 4 + q) * 16, 0, lane);

    for (int k0 = 0; k0 < Hn; k0 += 64) {
      // ping buffer <- chunk k0+32 (Hn % 64 == 0)
      v16h an = load_frag(h, Hn, mt * 16, k0 + 32, lane);
      v16h bn[4];
#pragma unroll
      for (int q = 0; q < 4; ++q) bn[q] = load_frag(Whh, Hn, (ngi * 4 + q) * 16, k0 + 32, lane);
#pragma unroll
      for (int q = 0; q < 4; ++q) acc[q] = wmma_f16(a, b[q], acc[q]);
      if (k0 + 64 < Hn) {
        a = load_frag(h, Hn, mt * 16, k0 + 64, lane);
#pragma unroll
        for (int q = 0; q < 4; ++q) b[q] = load_frag(Whh, Hn, (ngi * 4 + q) * 16, k0 + 64, lane);
      }
#pragma unroll
      for (int q = 0; q < 4; ++q) acc[q] = wmma_f16(an, bn[q], acc[q]);
    }
    {
      int mbase = mt * 16 + ((lane & 16) ? 8 : 0);
#pragma unroll
      for (int q = 0; q < 4; ++q) {
        int n = (ngi * 4 + q) * 16 + (lane & 15);
#pragma unroll
        for (int j = 0; j < 8; ++j) {
          int b_ = mbase + j;
          gates[(size_t)b_ * NG + n] = acc[q][j] + (float)xpt[(size_t)b_ * NG + n];
        }
      }
    }
    grid_barrier(gen, cnt, LSTM_WGS);

    for (int e = tid; e < Bn * Hn; e += LSTM_WGS * 256) {
      int b_ = e >> 8, hh = e & 255;
      const float* g_ = gates + (size_t)b_ * NG;
      float ig = sigmoidf_(g_[hh]);
      float fg = sigmoidf_(g_[Hn + hh]);
      float gg = tanhf(g_[2 * Hn + hh]);
      float og = sigmoidf_(g_[3 * Hn + hh]);
      float cn = fg * c[e] + ig * gg;
      float hn = og * tanhf(cn);
      c[e] = cn;
      h[e] = (_Float16)hn;
      if (hs) hs[((size_t)t * Bn + b_) * Hn + hh] = (_Float16)hn;
    }
    grid_barrier(gen, cnt, LSTM_WGS);
  }
  if (hT) {
    for (int e = tid; e < Bn * Hn; e += LSTM_WGS * 256) {
      int b_ = e >> 8, hh = e & 255;
      hT[(size_t)b_ * ldT + hh] = h[e];
    }
  }
}

// ---------------------------------------------------------------------------
// z = mean + eps*exp(0.5*logvar); nearest-of-16 VQ per latent dim; idx out;
// z_q as f16 for the l2h GEMM; deterministic two-stage vq_loss reduction.
// ---------------------------------------------------------------------------
__global__ __launch_bounds__(256) void latent_kernel(
    const float* __restrict__ mean, const float* __restrict__ logvar,
    const float* __restrict__ eps, const float* __restrict__ cb,
    int* __restrict__ idx_out, _Float16* __restrict__ zq_h,
    float* __restrict__ partials) {
  int i = blockIdx.x * 256 + threadIdx.x;       // i = b*L + l
  int l = i & (Ln - 1);
  float z = mean[i] + eps[i] * __expf(0.5f * logvar[i]);
  const float* cbl = cb + l * Kc;
  int best = 0;
  float bd = (z - cbl[0]) * (z - cbl[0]);
#pragma unroll
  for (int k = 1; k < Kc; ++k) {
    float d = z - cbl[k]; d *= d;
    if (d < bd) { bd = d; best = k; }
  }
  idx_out[i] = best;
  float zq = cbl[best];
  zq_h[i] = (_Float16)zq;
  float diff = zq - z;
  __shared__ float red[256];
  red[threadIdx.x] = diff * diff;
  __syncthreads();
  for (int s = 128; s > 0; s >>= 1) {
    if (threadIdx.x < s) red[threadIdx.x] += red[threadIdx.x + s];
    __syncthreads();
  }
  if (threadIdx.x == 0) partials[blockIdx.x] = red[0];
}

__global__ void vq_finish_kernel(const float* __restrict__ partials, float* __restrict__ out) {
  if (threadIdx.x == 0) {
    float s = 0.0f;
    for (int i = 0; i < (Bn * Ln) / 256; ++i) s += partials[i];
    out[0] = 1.25f * s / (float)Bn;   // q_loss + 0.25*e_loss, mean over batch
  }
}

// ---------------------------------------------------------------------------
extern "C" void kernel_launch(void* const* d_in, const int* in_sizes, int n_in,
                              void* d_out, int out_size, void* d_ws, size_t ws_size,
                              hipStream_t stream) {
  const int*   x        = (const int*)  d_in[0];
  const float* eps      = (const float*)d_in[1];
  const float* tok_emb  = (const float*)d_in[2];
  const float* Wih_f    = (const float*)d_in[3];
  const float* Whh_f    = (const float*)d_in[4];
  const float* b_f      = (const float*)d_in[5];
  const float* Wih_b    = (const float*)d_in[6];
  const float* Whh_b    = (const float*)d_in[7];
  const float* b_b      = (const float*)d_in[8];
  const float* W_mean   = (const float*)d_in[9];
  const float* b_mean   = (const float*)d_in[10];
  const float* W_logvar = (const float*)d_in[11];
  const float* b_logvar = (const float*)d_in[12];
  const float* cb       = (const float*)d_in[13];
  const float* W_l2h    = (const float*)d_in[14];
  const float* b_l2h    = (const float*)d_in[15];
  const float* Wih_d    = (const float*)d_in[16];
  const float* Whh_d    = (const float*)d_in[17];
  const float* b_d      = (const float*)d_in[18];
  const float* W_out    = (const float*)d_in[19];
  const float* b_out    = (const float*)d_in[20];

  float* out = (float*)d_out;

  // ---- workspace bump allocator (everything is L2-resident by design) ----
  char* p = (char*)d_ws;
  auto alloc = [&](size_t bytes) -> void* {
    void* r = (void*)p;
    p += (bytes + 255) & ~(size_t)255;
    return r;
  };
  unsigned* bar = (unsigned*)alloc(256);
  _Float16* emb   = (_Float16*)alloc((size_t)Tn * Bn * En * 2);
  _Float16* xp_f  = (_Float16*)alloc((size_t)Tn * Bn * NG * 2);
  _Float16* xp_b  = (_Float16*)alloc((size_t)Tn * Bn * NG * 2);
  _Float16* xp_d  = (_Float16*)alloc((size_t)Bn * NG * 2);
  float*    gates = (float*)   alloc((size_t)Bn * NG * 4);
  _Float16* hbuf  = (_Float16*)alloc((size_t)Bn * Hn * 2);
  float*    cbuf  = (float*)   alloc((size_t)Bn * Hn * 4);
  _Float16* hs_d  = (_Float16*)alloc((size_t)Tn * Bn * Hn * 2);
  _Float16* hcat  = (_Float16*)alloc((size_t)Bn * 2 * Hn * 2);
  _Float16* zq_h  = (_Float16*)alloc((size_t)Bn * Ln * 2);
  _Float16* hid_h = (_Float16*)alloc((size_t)Bn * Hn * 2);
  float*    parts = (float*)   alloc(((size_t)Bn * Ln / 256) * 4);
  _Float16* Wih_f_h  = (_Float16*)alloc((size_t)NG * En * 2);
  _Float16* Whh_f_h  = (_Float16*)alloc((size_t)NG * Hn * 2);
  _Float16* Wih_b_h  = (_Float16*)alloc((size_t)NG * En * 2);
  _Float16* Whh_b_h  = (_Float16*)alloc((size_t)NG * Hn * 2);
  _Float16* W_mean_h = (_Float16*)alloc((size_t)Ln * 2 * Hn * 2);
  _Float16* W_lv_h   = (_Float16*)alloc((size_t)Ln * 2 * Hn * 2);
  _Float16* W_l2h_h  = (_Float16*)alloc((size_t)Hn * Ln * 2);
  _Float16* Wih_d_h  = (_Float16*)alloc((size_t)NG * Hn * 2);
  _Float16* Whh_d_h  = (_Float16*)alloc((size_t)NG * Hn * 2);
  _Float16* W_out_h  = (_Float16*)alloc((size_t)Vocab * Hn * 2);
  (void)ws_size; (void)in_sizes; (void)n_in; (void)out_size;

  init_kernel<<<1, 64, 0, stream>>>(bar);

  auto cast = [&](const float* s, _Float16* d, int n) {
    cast_f16_kernel<<<(n + 255) / 256, 256, 0, stream>>>(s, d, n);
  };
  cast(Wih_f, Wih_f_h, NG * En);     cast(Whh_f, Whh_f_h, NG * Hn);
  cast(Wih_b, Wih_b_h, NG * En);     cast(Whh_b, Whh_b_h, NG * Hn);
  cast(W_mean, W_mean_h, Ln * 2 * Hn); cast(W_logvar, W_lv_h, Ln * 2 * Hn);
  cast(W_l2h, W_l2h_h, Hn * Ln);
  cast(Wih_d, Wih_d_h, NG * Hn);     cast(Whh_d, Whh_d_h, NG * Hn);
  cast(W_out, W_out_h, Vocab * Hn);

  embed_kernel<<<(Tn * Bn * En) / 256, 256, 0, stream>>>(x, tok_emb, emb);

  auto gemm = [&](const _Float16* A, const _Float16* W, const float* bias,
                  float* Cf, _Float16* Ch, int M, int N, int K, long ldc, int remapB) {
    dim3 grid(N / 64, (M + 127) / 128);
    gemm_kernel<<<grid, 256, 0, stream>>>(A, W, bias, Cf, Ch, M, N, K, ldc, remapB);
  };

  // xp precompute for both encoders: [T*B, 4H] = emb[T*B, E] @ Wih^T + b
  gemm(emb, Wih_f_h, b_f, nullptr, xp_f, Tn * Bn, NG, En, NG, 0);
  gemm(emb, Wih_b_h, b_b, nullptr, xp_b, Tn * Bn, NG, En, NG, 0);

  // encoder LSTMs (persistent, grid-synced); final h lands in hcat slices
  lstm_kernel<<<LSTM_WGS, 256, 0, stream>>>(xp_f, (long)Bn * NG, 0, Whh_f_h,
                                            hbuf, cbuf, gates, nullptr, hcat, 2 * Hn, bar);
  lstm_kernel<<<LSTM_WGS, 256, 0, stream>>>(xp_b, (long)Bn * NG, 1, Whh_b_h,
                                            hbuf, cbuf, gates, nullptr, hcat + Hn, 2 * Hn, bar);

  // mean / logvar straight into d_out
  gemm(hcat, W_mean_h, b_mean, out + MEAN_OFF, nullptr, Bn, Ln, 2 * Hn, Ln, 0);
  gemm(hcat, W_lv_h, b_logvar, out + LOGVAR_OFF, nullptr, Bn, Ln, 2 * Hn, Ln, 0);

  // reparam + VQ + losses
  latent_kernel<<<(Bn * Ln) / 256, 256, 0, stream>>>(
      out + MEAN_OFF, out + LOGVAR_OFF, eps, cb,
      (int*)(out + IDX_OFF), zq_h, parts);
  vq_finish_kernel<<<1, 32, 0, stream>>>(parts, out + VQ_OFF);

  // hid = z_q @ W_l2h^T + b ; xp_d = hid @ Wih_d^T + b (constant across T)
  gemm(zq_h, W_l2h_h, b_l2h, nullptr, hid_h, Bn, Hn, Ln, Hn, 0);
  gemm(hid_h, Wih_d_h, b_d, nullptr, xp_d, Bn, NG, Hn, NG, 0);

  // decoder LSTM, keeping all hidden states [T,B,H] in f16
  lstm_kernel<<<LSTM_WGS, 256, 0, stream>>>(xp_d, 0L, 0, Whh_d_h,
                                            hbuf, cbuf, gates, hs_d, nullptr, 0, bar);

  // logits[B,T,V] = hs_d[T*B,H] @ W_out^T + b_out  (row remap [T,B]->[B,T])
  gemm(hs_d, W_out_h, b_out, out, nullptr, Tn * Bn, Vocab, Hn, Vocab, Bn);
}
